// TagLoss_51135880626478
// MI455X (gfx1250) — matrix operand order repaired and checked by reading, not compiled, not run on hardware
//
#include <hip/hip_runtime.h>
#include <hip/hip_bf16.h>

typedef __attribute__((ext_vector_type(16))) _Float16 v16h;
typedef __attribute__((ext_vector_type(8)))  float    v8f;

#define NB 32    // batches
#define ND 64    // embedding dim
#define NH 192   // height
#define NW 128   // width
#define NK 64    // keypoints
#define ROWPAD 72  // padded LDS row length (halfs) to stagger banks

// One workgroup per batch: gather -> LDS(f16) -> 16 waves x (16x16 tile of G)
// via v_wmma_f32_16x16x32_f16, then loss terms + fixed-order tree reduction.
__global__ __launch_bounds__(512)
void TagLoss_batch_kernel(const float* __restrict__ ebd,
                          const float* __restrict__ kpts,
                          const int*   __restrict__ tags,
                          float*       __restrict__ batch_loss) {
  __shared__ _Float16 Ef[NK * ROWPAD];   // E[k][d] as f16
  __shared__ float    snorm[NK];         // ||e_k||^2 (f32, from f16 values)
  __shared__ int      stag[NK];
  __shared__ int      srow[NK];
  __shared__ int      scol[NK];
  __shared__ float    red[512];

  const int b   = blockIdx.x;
  const int tid = threadIdx.x;

  // --- keypoint -> pixel index, tags ---
  if (tid < NK) {
    float y = kpts[(b * NK + tid) * 2 + 0];
    float x = kpts[(b * NK + tid) * 2 + 1];
    int r = (int)floorf(y * (float)NH);
    int c = (int)floorf(x * (float)NW);
    r = r < 0 ? 0 : (r >= NH ? NH - 1 : r);
    c = c < 0 ? 0 : (c >= NW ? NW - 1 : c);
    srow[tid] = r;
    scol[tid] = c;
    stag[tid] = tags[b * NK + tid];
  }
  __syncthreads();

  // --- gather 64x64 embedding block into LDS as f16 (8 elems/thread) ---
#pragma unroll
  for (int e = 0; e < 8; ++e) {
    int idx = tid + e * 512;          // 0..4095
    int k = idx >> 6;                 // keypoint
    int d = idx & 63;                 // channel
    size_t off = (((size_t)b * ND + d) * NH + srow[k]) * NW + scol[k];
    Ef[k * ROWPAD + d] = (_Float16)ebd[off];
  }
  __syncthreads();

  // --- squared norms (consistent with WMMA's f16 inputs / f32 accum) ---
  if (tid < NK) {
    float s = 0.0f;
    const _Float16* row = &Ef[tid * ROWPAD];
#pragma unroll 8
    for (int d = 0; d < ND; ++d) {
      float v = (float)row[d];
      s += v * v;
    }
    snorm[tid] = s;
  }
  __syncthreads();

  // --- 16 waves: wave w owns tile (ti,tj) of the 64x64 Gram matrix ---
  const int wave = tid >> 5;
  const int lane = tid & 31;
  const int ti   = wave >> 2;
  const int tj   = wave & 3;
  const int g    = lane >> 4;   // lane half-group
  const int ml   = lane & 15;

  v8f acc = {};
#pragma unroll
  for (int kb = 0; kb < 2; ++kb) {
    const int kbase = kb * 32;
    v16h a, bm;
    // A fragment (16x32 f16, contiguous k order as for B): lane holds row m=ml;
    // half h -> k = h + 16*g
    const _Float16* arow = &Ef[(ti * 16 + ml) * ROWPAD + kbase];
#pragma unroll
    for (int h = 0; h < 16; ++h) {
      int kk = h + 16 * g;
      a[h] = arow[kk];
    }
    // B fragment (32x16 f16): B[k][n] = E[jbase+n][k]; lane -> col n=ml,
    // half h -> k = h + 16*g (contiguous 32B per lane)
    const _Float16* brow = &Ef[(tj * 16 + ml) * ROWPAD + kbase + 16 * g];
#pragma unroll
    for (int h = 0; h < 16; ++h) {
      bm[h] = brow[h];
    }
    acc = __builtin_amdgcn_wmma_f32_16x16x32_f16(
        /*neg_a=*/false, a, /*neg_b=*/false, bm,
        /*c_mod=*/(short)0, acc, /*reuse_a=*/false, /*reuse_b=*/false);
  }

  // --- C/D layout: VGPR r -> M = r + 8*g, N = ml ---
  float partial = 0.0f;
#pragma unroll
  for (int r = 0; r < 8; ++r) {
    int i = ti * 16 + r + 8 * g;
    int j = tj * 16 + ml;
    float gij  = acc[r];
    float expo = (snorm[i] + snorm[j] - 2.0f * gij) * (1.0f / (float)ND);
    // pred = 2 / (1 + e^expo), with fast hardware rcp instead of IEEE divide
    float denom = 1.0f + __expf(expo);
    float pred  = 2.0f * __builtin_amdgcn_rcpf(denom);
    float tsim = (stag[i] == stag[j]) ? 1.0f : 0.0f;
    float df   = pred - tsim;
    partial += df * df;
  }

  // --- deterministic block tree reduction ---
  red[tid] = partial;
  __syncthreads();
  for (int s = 256; s > 0; s >>= 1) {
    if (tid < s) red[tid] += red[tid + s];
    __syncthreads();
  }
  if (tid == 0) batch_loss[b] = red[0] * (1.0f / (float)(NK * NK));
}

// Fixed-order final mean over batches -> scalar (deterministic, no atomics).
__global__ void TagLoss_final_kernel(const float* __restrict__ batch_loss,
                                     float* __restrict__ out) {
  if (threadIdx.x == 0 && blockIdx.x == 0) {
    float s = 0.0f;
    for (int b = 0; b < NB; ++b) s += batch_loss[b];
    out[0] = s * (1.0f / (float)NB);
  }
}

extern "C" void kernel_launch(void* const* d_in, const int* in_sizes, int n_in,
                              void* d_out, int out_size, void* d_ws, size_t ws_size,
                              hipStream_t stream) {
  const float* ebd  = (const float*)d_in[0];  // [B, D, H, W] f32
  const float* kpts = (const float*)d_in[1];  // [B, K, 2]   f32
  const int*   tags = (const int*)d_in[2];    // [B, K]      int
  float* out = (float*)d_out;                 // scalar f32
  float* bl  = (float*)d_ws;                  // 32 floats scratch

  TagLoss_batch_kernel<<<NB, 512, 0, stream>>>(ebd, kpts, tags, bl);
  TagLoss_final_kernel<<<1, 32, 0, stream>>>(bl, out);
}
